// GNN_49134425866246
// MI455X (gfx1250) — matrix operand order repaired and checked by reading, not compiled, not run on hardware
//
#include <hip/hip_runtime.h>

// GraphSAGE 3-layer forward for MI455X (gfx1250, wave32, WMMA).
// Aggregation is HBM/atomic-bound (~5 GB streaming) -> plain f32 atomics +
// global_prefetch for the gather stream. Dense math uses
// v_wmma_f32_16x16x32_bf16 with f32 accumulation; weights staged in LDS so
// B-fragments are ds_load_b128 instead of serialized global loads.

#define D 128                    // feature dim (in == hidden)
#define COUT 40                  // output classes
#define CPAD 48                  // padded to 3 x 16 col tiles

typedef __attribute__((ext_vector_type(16))) __bf16 v16bf;
typedef __attribute__((ext_vector_type(8)))  float  v8f;

union BfVec { unsigned short u[16]; v16bf v; };

__device__ __forceinline__ unsigned short f2bf(float f) {
    unsigned u = __builtin_bit_cast(unsigned, f);
    return (unsigned short)((u + 0x7FFFu + ((u >> 16) & 1u)) >> 16);  // RNE
}

// ---------------------------------------------------------------- utilities
__global__ void fill_zero_kernel(float* __restrict__ p, long long n) {
    long long i = (long long)blockIdx.x * blockDim.x + threadIdx.x;
    long long stride = (long long)gridDim.x * blockDim.x;
    for (; i < n; i += stride) p[i] = 0.0f;
}

__global__ void degree_kernel(const int* __restrict__ dst, float* __restrict__ cnt, int E) {
    int e = blockIdx.x * blockDim.x + threadIdx.x;
    if (e < E) atomicAdd(&cnt[dst[e]], 1.0f);
}

__global__ void invcnt_kernel(const float* __restrict__ cnt, float* __restrict__ inv, int N) {
    int i = blockIdx.x * blockDim.x + threadIdx.x;
    if (i < N) inv[i] = 1.0f / fmaxf(cnt[i], 1.0f);
}

__global__ void convert_weights_kernel(const float* __restrict__ Wl,
                                       const float* __restrict__ Wr,
                                       const float* __restrict__ Wlin,
                                       unsigned short* __restrict__ Wlb,
                                       unsigned short* __restrict__ Wrb,
                                       unsigned short* __restrict__ Wlinb) {
    int i = blockIdx.x * blockDim.x + threadIdx.x;
    const int totW = 3 * D * D;
    if (i < totW) { Wlb[i] = f2bf(Wl[i]); Wrb[i] = f2bf(Wr[i]); }
    if (i < CPAD * D) {
        int r = i / D, c = i - r * D;
        Wlinb[i] = (r < COUT) ? f2bf(Wlin[r * D + c]) : (unsigned short)0;
    }
}

// One wave (32 lanes) per edge; each lane moves 4 contiguous floats.
// Prefetch the gather row 64 edges ahead to deepen the HBM pipeline.
__global__ void scatter_kernel(const float* __restrict__ z,
                               const int* __restrict__ src,
                               const int* __restrict__ dst,
                               float* __restrict__ agg, int E) {
    int idx = blockIdx.x * blockDim.x + threadIdx.x;
    if (idx >= E * 32) return;
    int e  = idx >> 5;
    int c4 = (idx & 31) << 2;
    if ((idx & 31) == 0 && e + 64 < E) {
        __builtin_prefetch(z + (size_t)src[e + 64] * D, 0, 0);  // global_prefetch_b8
    }
    int s = src[e], d = dst[e];
    const float4 v = *reinterpret_cast<const float4*>(z + (size_t)s * D + c4);
    float* a = agg + (size_t)d * D + c4;
    atomicAdd(a + 0, v.x);
    atomicAdd(a + 1, v.y);
    atomicAdd(a + 2, v.z);
    atomicAdd(a + 3, v.w);
}

// ------------------------------------------------- A-fragment construction
// 16-bit A layout (16x32), contiguous to match B: lane (m + 16*h) holds row m with
//   elements e=0..15 <-> K = kb + 16*h + e
__device__ __forceinline__ v16bf load_a_frag(const float* __restrict__ base,
                                             int kb, int half, float scale) {
    BfVec t;
#pragma unroll
    for (int g = 0; g < 2; ++g) {
        const float4* p = reinterpret_cast<const float4*>(base + kb + 16 * half + 8 * g);
        float4 f0 = p[0], f1 = p[1];
        t.u[8 * g + 0] = f2bf(f0.x * scale);
        t.u[8 * g + 1] = f2bf(f0.y * scale);
        t.u[8 * g + 2] = f2bf(f0.z * scale);
        t.u[8 * g + 3] = f2bf(f0.w * scale);
        t.u[8 * g + 4] = f2bf(f1.x * scale);
        t.u[8 * g + 5] = f2bf(f1.y * scale);
        t.u[8 * g + 6] = f2bf(f1.z * scale);
        t.u[8 * g + 7] = f2bf(f1.w * scale);
    }
    return t.v;
}

// B = W^T: column n of B is row n of W; lane (n + 16*h) reads 16 contiguous
// bf16 of W[n][kc*32 + 16*h ..] -> two ds_load_b128 from LDS.
__device__ __forceinline__ v16bf load_b_frag_lds(const unsigned short* sW,
                                                 int col, int half, int kc) {
    return *reinterpret_cast<const v16bf*>(sW + (size_t)col * D + 16 * half + kc * 32);
}

// ------------------------------------------------------------- SAGE layer
// out = L2norm_relu( agg@Wl^T + bl + z@Wr^T ), one 16x128 tile per wave,
// 8 waves per block, both weight matrices staged in LDS (64 KB dynamic).
__global__ __launch_bounds__(256) void sage_layer_kernel(
    const float* __restrict__ agg, const float* __restrict__ inv,
    const float* __restrict__ zin, const unsigned short* __restrict__ Wlb,
    const float* __restrict__ bl, const unsigned short* __restrict__ Wrb,
    float* __restrict__ zout, int N, int ntiles) {
    extern __shared__ unsigned short smem[];        // [2][D*D] bf16
    unsigned short* sWl = smem;
    unsigned short* sWr = smem + D * D;

    {   // cooperative stage: 2 x 32 KB, uint4 (8 bf16) granularity
        const uint4* gl = reinterpret_cast<const uint4*>(Wlb);
        const uint4* gr = reinterpret_cast<const uint4*>(Wrb);
        uint4* sl = reinterpret_cast<uint4*>(sWl);
        uint4* sr = reinterpret_cast<uint4*>(sWr);
        const int nvec = D * D / 8;                 // 2048
        for (int i = threadIdx.x; i < nvec; i += blockDim.x) {
            sl[i] = gl[i];
            sr[i] = gr[i];
        }
    }
    __syncthreads();

    int wave = threadIdx.x >> 5;
    int lane = threadIdx.x & 31;
    int tile = blockIdx.x * (blockDim.x >> 5) + wave;
    if (tile >= ntiles) return;
    int m = lane & 15;       // A row / B,C column within tile
    int half = lane >> 4;

    int arow = tile * 16 + m;
    if (arow > N - 1) arow = N - 1;                 // tail clamp (harmless dup)
    float mean_scale = inv[arow];                   // fused mean aggregation

    const float* aggrow = agg + (size_t)arow * D;
    const float* zrow   = zin + (size_t)arow * D;
    v16bf aA[4], aZ[4];
#pragma unroll
    for (int kc = 0; kc < 4; ++kc) {
        aA[kc] = load_a_frag(aggrow, kc * 32, half, mean_scale);
        aZ[kc] = load_a_frag(zrow,   kc * 32, half, 1.0f);
    }

    v8f c[8];
#pragma unroll
    for (int j = 0; j < 8; ++j) {
        float b = bl[j * 16 + m];                   // bias: same col for all 8 rows
#pragma unroll
        for (int r = 0; r < 8; ++r) c[j][r] = b;
    }

#pragma unroll
    for (int j = 0; j < 8; ++j) {
        int col = j * 16 + m;
#pragma unroll
        for (int kc = 0; kc < 4; ++kc) {
            v16bf bL = load_b_frag_lds(sWl, col, half, kc);
            c[j] = __builtin_amdgcn_wmma_f32_16x16x32_bf16(
                false, aA[kc], false, bL, (short)0, c[j], false, false);
            v16bf bR = load_b_frag_lds(sWr, col, half, kc);
            c[j] = __builtin_amdgcn_wmma_f32_16x16x32_bf16(
                false, aZ[kc], false, bR, (short)0, c[j], false, false);
        }
    }

    // Row L2 norm: C layout -> lanes 0-15 hold rows r, lanes 16-31 rows r+8;
    // xor-masks < 16 reduce within each half independently.
    float s[8];
#pragma unroll
    for (int r = 0; r < 8; ++r) s[r] = 0.0f;
#pragma unroll
    for (int j = 0; j < 8; ++j)
#pragma unroll
        for (int r = 0; r < 8; ++r) { float v = c[j][r]; s[r] += v * v; }
#pragma unroll
    for (int r = 0; r < 8; ++r) {
        float v = s[r];
        v += __shfl_xor(v, 1, 32);
        v += __shfl_xor(v, 2, 32);
        v += __shfl_xor(v, 4, 32);
        v += __shfl_xor(v, 8, 32);
        s[r] = 1.0f / fmaxf(sqrtf(v), 1e-12f);
    }
#pragma unroll
    for (int j = 0; j < 8; ++j)
#pragma unroll
        for (int r = 0; r < 8; ++r) {
            int grow = tile * 16 + r + 8 * half;
            if (grow < N) {
                float v = fmaxf(c[j][r] * s[r], 0.0f);  // relu(x*pos_scale)
                zout[(size_t)grow * D + j * 16 + m] = v;
            }
        }
}

// --------------------------------------------------------- final linear
__global__ __launch_bounds__(256) void final_linear_kernel(
    const float* __restrict__ zin, const unsigned short* __restrict__ Wlinb,
    const float* __restrict__ blin, float* __restrict__ out, int N, int ntiles) {
    extern __shared__ unsigned short smemf[];       // [CPAD*D] bf16 (12 KB)
    {
        const uint4* g = reinterpret_cast<const uint4*>(Wlinb);
        uint4* sp = reinterpret_cast<uint4*>(smemf);
        const int nvec = CPAD * D / 8;              // 768
        for (int i = threadIdx.x; i < nvec; i += blockDim.x) sp[i] = g[i];
    }
    __syncthreads();

    int wave = threadIdx.x >> 5;
    int lane = threadIdx.x & 31;
    int tile = blockIdx.x * (blockDim.x >> 5) + wave;
    if (tile >= ntiles) return;
    int m = lane & 15;
    int half = lane >> 4;

    int arow = tile * 16 + m;
    if (arow > N - 1) arow = N - 1;
    const float* zrow = zin + (size_t)arow * D;
    v16bf aZ[4];
#pragma unroll
    for (int kc = 0; kc < 4; ++kc) aZ[kc] = load_a_frag(zrow, kc * 32, half, 1.0f);

    v8f c[3];
#pragma unroll
    for (int j = 0; j < 3; ++j) {
        int col = j * 16 + m;
        float b = (col < COUT) ? blin[col] : 0.0f;
#pragma unroll
        for (int r = 0; r < 8; ++r) c[j][r] = b;
    }
#pragma unroll
    for (int j = 0; j < 3; ++j) {
        int col = j * 16 + m;
#pragma unroll
        for (int kc = 0; kc < 4; ++kc) {
            v16bf bF = load_b_frag_lds(smemf, col, half, kc);
            c[j] = __builtin_amdgcn_wmma_f32_16x16x32_bf16(
                false, aZ[kc], false, bF, (short)0, c[j], false, false);
        }
    }
#pragma unroll
    for (int j = 0; j < 3; ++j) {
        int col = j * 16 + m;
        if (col >= COUT) continue;
#pragma unroll
        for (int r = 0; r < 8; ++r) {
            int grow = tile * 16 + r + 8 * half;
            if (grow < N) out[(size_t)grow * COUT + col] = c[j][r];
        }
    }
}

// ------------------------------------------------------------------ launch
extern "C" void kernel_launch(void* const* d_in, const int* in_sizes, int n_in,
                              void* d_out, int out_size, void* d_ws, size_t ws_size,
                              hipStream_t stream) {
    const float* x    = (const float*)d_in[0];
    const int*   ei   = (const int*)d_in[1];   // [2, E]: src then dst
    const float* Wl   = (const float*)d_in[2]; // [3,128,128]
    const float* bl   = (const float*)d_in[3]; // [3,128]
    const float* Wr   = (const float*)d_in[4]; // [3,128,128]
    const float* Wlin = (const float*)d_in[5]; // [40,128]
    const float* blin = (const float*)d_in[6]; // [40]
    float* out = (float*)d_out;

    const int N = in_sizes[0] / D;
    const int E = in_sizes[1] / 2;
    const int* src = ei;
    const int* dst = ei + E;
    const int ntiles = (N + 15) / 16;

    char* ws = (char*)d_ws;
    size_t off = 0;
    auto alloc = [&](size_t bytes) -> void* {
        off = (off + 255) & ~(size_t)255;
        void* p = ws + off;
        off += bytes;
        return p;
    };
    float* cnt = (float*)alloc((size_t)N * 4);
    float* inv = (float*)alloc((size_t)N * 4);
    float* agg = (float*)alloc((size_t)N * D * 4);
    float* z1  = (float*)alloc((size_t)N * D * 4);
    float* z2  = (float*)alloc((size_t)N * D * 4);
    unsigned short* Wlb   = (unsigned short*)alloc((size_t)3 * D * D * 2);
    unsigned short* Wrb   = (unsigned short*)alloc((size_t)3 * D * D * 2);
    unsigned short* Wlinb = (unsigned short*)alloc((size_t)CPAD * D * 2);
    (void)ws_size; (void)n_in; (void)out_size;

    // degree -> 1/max(cnt,1), weight bf16 conversion
    fill_zero_kernel<<<1024, 256, 0, stream>>>(cnt, (long long)N);
    degree_kernel<<<(E + 255) / 256, 256, 0, stream>>>(dst, cnt, E);
    invcnt_kernel<<<(N + 255) / 256, 256, 0, stream>>>(cnt, inv, N);
    convert_weights_kernel<<<(3 * D * D + 255) / 256, 256, 0, stream>>>(
        Wl, Wr, Wlin, Wlb, Wrb, Wlinb);

    const float* zp = x;
    float* zbuf[2] = { z1, z2 };
    const int gemm_blocks = (ntiles + 7) / 8;        // 8 waves (tiles) per block
    const size_t lds_layer = (size_t)2 * D * D * 2;  // 64 KB
    const size_t lds_final = (size_t)CPAD * D * 2;   // 12 KB
    for (int layer = 0; layer < 3; ++layer) {
        fill_zero_kernel<<<4096, 256, 0, stream>>>(agg, (long long)N * D);
        int sthreads = E * 32;
        scatter_kernel<<<(sthreads + 255) / 256, 256, 0, stream>>>(zp, src, dst, agg, E);
        float* znext = zbuf[layer & 1];
        sage_layer_kernel<<<gemm_blocks, 256, lds_layer, stream>>>(
            agg, inv, zp,
            Wlb + (size_t)layer * D * D, bl + (size_t)layer * D,
            Wrb + (size_t)layer * D * D, znext, N, ntiles);
        zp = znext;
    }
    final_linear_kernel<<<gemm_blocks, 256, lds_final, stream>>>(zp, Wlinb, blin, out, N, ntiles);
}
